// GNNEncoder_48266842472715
// MI455X (gfx1250) — matrix operand order repaired and checked by reading, not compiled, not run on hardware
//
#include <hip/hip_runtime.h>

#define N_NODES  50000
#define N_EDGES  600000
#define N_GRAPHS 64
#define IN_D     5
#define HID      128
#define EMB      64

typedef __attribute__((ext_vector_type(16))) _Float16 v16h;
typedef __attribute__((ext_vector_type(8)))  float    v8f;

// ---------------- small utility kernels ----------------

__global__ __launch_bounds__(256) void k_fill(float* __restrict__ p, float v, int n) {
  int i = blockIdx.x * 256 + threadIdx.x;
  if (i < n) p[i] = v;
}

__global__ __launch_bounds__(256) void k_deg_scatter(const int* __restrict__ dst,
                                                     float* __restrict__ deg, int ne) {
  int e = blockIdx.x * 256 + threadIdx.x;
  if (e < ne) atomicAdd(&deg[dst[e]], 1.0f);
}

__global__ __launch_bounds__(256) void k_rsqrt_inplace(float* __restrict__ p, int n) {
  int i = blockIdx.x * 256 + threadIdx.x;
  if (i < n) p[i] = rsqrtf(p[i]);   // deg >= 1 (self loop) so always valid
}

// ---------------- layer-1 GEMM: x(N x 5) @ W1(5 x 128) ----------------

__global__ __launch_bounds__(256) void k_gemm_in(const float* __restrict__ x,
                                                 const float* __restrict__ W,
                                                 float* __restrict__ HW, int total) {
  int t = blockIdx.x * 256 + threadIdx.x;
  if (t >= total) return;
  int i = t >> 7, c = t & 127;
  const float* xr = x + i * IN_D;
  float a = 0.f;
#pragma unroll
  for (int k = 0; k < IN_D; ++k) a += xr[k] * W[k * HID + c];
  HW[t] = a;
}

// ---------------- W (128x128 f32 row-major) -> Wt (f16 col-major) ----------------

__global__ __launch_bounds__(256) void k_cvt_w(const float* __restrict__ W,
                                               _Float16* __restrict__ Wt) {
  int t = blockIdx.x * 256 + threadIdx.x;   // 16384 threads
  int c = t >> 7, k = t & 127;
  Wt[t] = (_Float16)W[k * HID + c];         // Wt[c*128 + k]
}

// ---------------- WMMA GEMM: H(N x 128 f32) @ W(128 x 128) -> HW f32 ----------------
// block = 256 threads = 8 waves. Block owns 16 rows; wave w owns cols [16w, 16w+16).

__global__ __launch_bounds__(256) void k_gemm_wmma(const float* __restrict__ H,
                                                   const _Float16* __restrict__ Wt,
                                                   float* __restrict__ HW) {
  const int wave = threadIdx.x >> 5;
  const int lane = threadIdx.x & 31;
  const int h16  = lane >> 4;        // 0 or 1 (half-wave)
  const int l15  = lane & 15;
  const int rowBase = blockIdx.x << 4;
  const int arow = rowBase + l15;            // A-matrix row for this lane
  const int bcol = (wave << 4) + l15;        // B-matrix column for this lane

  v8f acc = {};
#pragma unroll
  for (int k0 = 0; k0 < HID; k0 += 32) {
    // A fragment: halves 0..15 -> K = k0 + 16*h16 + h
    const float* ap = H + arow * HID + k0 + (h16 << 4);
    float4 a0 = *(const float4*)(ap);
    float4 a1 = *(const float4*)(ap + 4);
    float4 a2 = *(const float4*)(ap + 8);
    float4 a3 = *(const float4*)(ap + 12);
    v16h A;
    A[0]=(_Float16)a0.x;  A[1]=(_Float16)a0.y;  A[2]=(_Float16)a0.z;  A[3]=(_Float16)a0.w;
    A[4]=(_Float16)a1.x;  A[5]=(_Float16)a1.y;  A[6]=(_Float16)a1.z;  A[7]=(_Float16)a1.w;
    A[8]=(_Float16)a2.x;  A[9]=(_Float16)a2.y;  A[10]=(_Float16)a2.z; A[11]=(_Float16)a2.w;
    A[12]=(_Float16)a3.x; A[13]=(_Float16)a3.y; A[14]=(_Float16)a3.z; A[15]=(_Float16)a3.w;
    // B fragment: lane holds col bcol, K = k0 + 16*h16 + h  (contiguous in col-major Wt)
    v16h B = *(const v16h*)(Wt + bcol * HID + k0 + (h16 << 4));
    acc = __builtin_amdgcn_wmma_f32_16x16x32_f16(false, A, false, B, (short)0, acc,
                                                 false, false);
  }
  // D layout: VGPR i -> row rowBase + 8*h16 + i, col = wave*16 + l15
#pragma unroll
  for (int i = 0; i < 8; ++i)
    HW[(rowBase + (h16 << 3) + i) * HID + (wave << 4) + l15] = acc[i];
}

// ---------------- self-loop init: AGG = HW * dinv^2 ----------------

__global__ __launch_bounds__(256) void k_selfinit(const float* __restrict__ HW,
                                                  const float* __restrict__ dinv,
                                                  float* __restrict__ AGG, int total) {
  int t = blockIdx.x * 256 + threadIdx.x;
  if (t >= total) return;
  float di = dinv[t >> 7];
  AGG[t] = HW[t] * di * di;
}

// ---------------- edge scatter: one wave per edge, 4 cols per lane ----------------

__global__ __launch_bounds__(256) void k_edge(const float* __restrict__ HW,
                                              const int* __restrict__ src,
                                              const int* __restrict__ dst,
                                              const float* __restrict__ dinv,
                                              float* __restrict__ AGG, int ne) {
  int e = blockIdx.x * 8 + (threadIdx.x >> 5);
  if (e >= ne) return;
  int lane = threadIdx.x & 31;
  int s = src[e], d = dst[e];
  float w = dinv[s] * dinv[d];
  float4 v = *(const float4*)(HW + s * HID + (lane << 2));
  float* o = AGG + d * HID + (lane << 2);
  atomicAdd(o + 0, v.x * w);
  atomicAdd(o + 1, v.y * w);
  atomicAdd(o + 2, v.z * w);
  atomicAdd(o + 3, v.w * w);
}

// ---------------- +bias, BatchNorm(eval), ReLU (in place) ----------------

__global__ __launch_bounds__(256) void k_bnrelu(float* __restrict__ A,
                                                const float* __restrict__ b,
                                                const float* __restrict__ g,
                                                const float* __restrict__ be, int total) {
  int t = blockIdx.x * 256 + threadIdx.x;
  if (t >= total) return;
  int c = t & 127;
  float s = g[c] * rsqrtf(1.0f + 1e-5f);
  float v = (A[t] + b[c]) * s + be[c];
  A[t] = v > 0.f ? v : 0.f;
}

// ---------------- global mean pool ----------------

__global__ __launch_bounds__(256) void k_pool(const float* __restrict__ H,
                                              const int* __restrict__ batch,
                                              float* __restrict__ pooled,
                                              float* __restrict__ cnt, int total) {
  int t = blockIdx.x * 256 + threadIdx.x;
  if (t >= total) return;
  int i = t >> 7, c = t & 127;
  int gph = batch[i];
  atomicAdd(&pooled[gph * HID + c], H[t]);
  if (c == 0) atomicAdd(&cnt[gph], 1.0f);
}

// ---------------- projection: (pooled/cnt) @ Wp + bp ----------------

__global__ __launch_bounds__(256) void k_proj(const float* __restrict__ pooled,
                                              const float* __restrict__ cnt,
                                              const float* __restrict__ Wp,
                                              const float* __restrict__ bp,
                                              float* __restrict__ out) {
  int t = blockIdx.x * 256 + threadIdx.x;   // 4096 threads
  if (t >= N_GRAPHS * EMB) return;
  int gph = t >> 6, j = t & 63;
  float ic = 1.0f / fmaxf(cnt[gph], 1.0f);
  float a = 0.f;
#pragma unroll 8
  for (int c = 0; c < HID; ++c) a += pooled[gph * HID + c] * Wp[c * EMB + j];
  out[t] = a * ic + bp[j];
}

// ---------------- host side ----------------

extern "C" void kernel_launch(void* const* d_in, const int* in_sizes, int n_in,
                              void* d_out, int out_size, void* d_ws, size_t ws_size,
                              hipStream_t stream) {
  (void)in_sizes; (void)n_in; (void)out_size; (void)ws_size;
  const float* x     = (const float*)d_in[0];
  const int*   src   = (const int*)d_in[1];
  const int*   dst   = (const int*)d_in[2];
  const int*   batch = (const int*)d_in[3];
  const float* W1 = (const float*)d_in[4];   const float* b1  = (const float*)d_in[5];
  const float* W2 = (const float*)d_in[6];   const float* b2  = (const float*)d_in[7];
  const float* W3 = (const float*)d_in[8];   const float* b3  = (const float*)d_in[9];
  const float* g1 = (const float*)d_in[10];  const float* be1 = (const float*)d_in[11];
  const float* g2 = (const float*)d_in[12];  const float* be2 = (const float*)d_in[13];
  const float* g3 = (const float*)d_in[14];  const float* be3 = (const float*)d_in[15];
  const float* Wp = (const float*)d_in[16];  const float* bp  = (const float*)d_in[17];
  float* out = (float*)d_out;

  char* ws = (char*)d_ws;
  size_t off = 0;
  auto carve = [&](size_t bytes) -> void* {
    void* p = ws + off;
    off = (off + bytes + 255) & ~(size_t)255;
    return p;
  };
  float*    bufA   = (float*)carve((size_t)N_NODES * HID * 4);  // H / AGG
  float*    bufB   = (float*)carve((size_t)N_NODES * HID * 4);  // HW
  float*    dinv   = (float*)carve((size_t)N_NODES * 4);
  _Float16* Wt     = (_Float16*)carve((size_t)HID * HID * 2);
  float*    pooled = (float*)carve((size_t)N_GRAPHS * HID * 4);
  float*    cnt    = (float*)carve((size_t)N_GRAPHS * 4);

  const int NC = N_NODES * HID;
  const dim3 blk(256);
  const int gNC    = (NC + 255) / 256;
  const int gNodes = (N_NODES + 255) / 256;
  const int gEdgeT = (N_EDGES + 255) / 256;
  const int gEdgeW = (N_EDGES + 7) / 8;        // 8 waves (edges) per block
  const int gGemm  = N_NODES / 16;             // 50000 / 16 = 3125

  // symmetric-norm degree factors (deg includes self loop)
  k_fill<<<gNodes, blk, 0, stream>>>(dinv, 1.0f, N_NODES);
  k_deg_scatter<<<gEdgeT, blk, 0, stream>>>(dst, dinv, N_EDGES);
  k_rsqrt_inplace<<<gNodes, blk, 0, stream>>>(dinv, N_NODES);

  // ---- layer 1 (K=5 GEMM, scalar) ----
  k_gemm_in<<<gNC, blk, 0, stream>>>(x, W1, bufB, NC);
  k_selfinit<<<gNC, blk, 0, stream>>>(bufB, dinv, bufA, NC);
  k_edge<<<gEdgeW, blk, 0, stream>>>(bufB, src, dst, dinv, bufA, N_EDGES);
  k_bnrelu<<<gNC, blk, 0, stream>>>(bufA, b1, g1, be1, NC);

  // ---- layer 2 (WMMA GEMM) ----
  k_cvt_w<<<(HID * HID + 255) / 256, blk, 0, stream>>>(W2, Wt);
  k_gemm_wmma<<<gGemm, blk, 0, stream>>>(bufA, Wt, bufB);
  k_selfinit<<<gNC, blk, 0, stream>>>(bufB, dinv, bufA, NC);
  k_edge<<<gEdgeW, blk, 0, stream>>>(bufB, src, dst, dinv, bufA, N_EDGES);
  k_bnrelu<<<gNC, blk, 0, stream>>>(bufA, b2, g2, be2, NC);

  // ---- layer 3 (WMMA GEMM) ----
  k_cvt_w<<<(HID * HID + 255) / 256, blk, 0, stream>>>(W3, Wt);
  k_gemm_wmma<<<gGemm, blk, 0, stream>>>(bufA, Wt, bufB);
  k_selfinit<<<gNC, blk, 0, stream>>>(bufB, dinv, bufA, NC);
  k_edge<<<gEdgeW, blk, 0, stream>>>(bufB, src, dst, dinv, bufA, N_EDGES);
  k_bnrelu<<<gNC, blk, 0, stream>>>(bufA, b3, g3, be3, NC);

  // ---- mean pool + projection ----
  k_fill<<<(N_GRAPHS * HID + 255) / 256, blk, 0, stream>>>(pooled, 0.f, N_GRAPHS * HID);
  k_fill<<<1, blk, 0, stream>>>(cnt, 0.f, N_GRAPHS);
  k_pool<<<gNC, blk, 0, stream>>>(bufA, batch, pooled, cnt, NC);
  k_proj<<<(N_GRAPHS * EMB + 255) / 256, blk, 0, stream>>>(pooled, cnt, Wp, bp, out);
}
